// MultiHeadedSelfAttention_48713519071502
// MI455X (gfx1250) — matrix operand order repaired and checked by reading, not compiled, not run on hardware
//
#include <hip/hip_runtime.h>
#include <hip/hip_bf16.h>

// ---------------------------------------------------------------------------
// MultiHeadedSelfAttention for MI455X (gfx1250, wave32, WMMA bf16)
//   B=4, C=256, S=4096, O=256, H=4, head dim WD=64
// Kernel 1: fused Q/K/V channel-mix projections (WMMA over K=C=256).
// Kernel 2: flash-style attention, softmax over the *s* axis per output
//           column t; online stats per wave + LDS cross-wave combine.
// attn_kernel WMMA fragment loads are 16B-aligned b128, contiguous k = 16*half + s order.
// ---------------------------------------------------------------------------

typedef __attribute__((ext_vector_type(16))) __bf16 v16bf;
typedef __attribute__((ext_vector_type(2)))  __bf16 v2bf;
typedef __attribute__((ext_vector_type(8)))  float  v8f;
typedef __attribute__((ext_vector_type(4)))  unsigned v4u;

union Frag {
  v16bf v;
  v4u   q[2];
  unsigned u[8];
  unsigned short h[16];
};

// two f32 -> packed bf16 pair (native cvt on gfx1250)
__device__ __forceinline__ unsigned pk2(float a, float b) {
  v2bf t;
  t[0] = (__bf16)a;
  t[1] = (__bf16)b;
  return __builtin_bit_cast(unsigned, t);
}
__device__ __forceinline__ unsigned short bf1(float a) {
  return __builtin_bit_cast(unsigned short, (__bf16)a);
}

constexpr int BB = 4, CC = 256, SS = 4096, OO = 256, HH = 4, WD = 64;

// ---------------------------------------------------------------------------
// Projection: one wave -> one 16(o) x 16(s) tile for q, k AND v
// (shares the X B-fragment across the three weight matrices).
//   Qtr[bh][s][w]  bf16, (Wq X + bq)/8      Ktr[bh][t][w]  bf16, Wk X + bk
//   Vb [bh][w][s]  bf16, Wv X + bv
// ---------------------------------------------------------------------------
__global__ __launch_bounds__(256)
void proj_kernel(const float* __restrict__ X,
                 const float* __restrict__ Wq, const float* __restrict__ bq,
                 const float* __restrict__ Wk, const float* __restrict__ bk,
                 const float* __restrict__ Wv, const float* __restrict__ bv,
                 unsigned short* __restrict__ Qtr,
                 unsigned short* __restrict__ Ktr,
                 unsigned short* __restrict__ Vb)
{
  const int tid  = threadIdx.x;
  const int wave = tid >> 5, lane = tid & 31;
  const int gw   = blockIdx.x * 8 + wave;          // 16384 waves total
  const int b    = gw >> 12;
  const int rem  = gw & 4095;
  const int o_base = (rem >> 8) * 16;
  const int s_base = (rem & 255) * 16;
  const int half = lane >> 4, l16 = lane & 15;
  const int s    = s_base + l16;                   // D/B column of this lane
  const int orow = o_base + l16;                   // A row of this lane

  v8f cq = {}, ck = {}, cv = {};
  for (int kk = 0; kk < 8; ++kk) {                 // K = C loop, 32 per step
    // B fragment from X: B[k=c, n=s]; K = kk*32 + half*16 + v*2 + pos
    Frag xb;
    #pragma unroll
    for (int v = 0; v < 8; ++v) {
      int c0 = kk * 32 + half * 16 + v * 2;
      float f0 = X[(b * CC + c0)     * SS + s];
      float f1 = X[(b * CC + c0 + 1) * SS + s];
      xb.u[v] = pk2(f0, f1);
    }
    // A fragments (rows = o, K pairs contiguous in c in row-major W)
    Frag aq, ak, av;
    #pragma unroll
    for (int v = 0; v < 8; ++v) {
      int c0 = kk * 32 + half * 16 + v * 2;
      float2 fq = *(const float2*)(Wq + orow * CC + c0);
      float2 fk = *(const float2*)(Wk + orow * CC + c0);
      float2 fv = *(const float2*)(Wv + orow * CC + c0);
      aq.u[v] = pk2(fq.x, fq.y);
      ak.u[v] = pk2(fk.x, fk.y);
      av.u[v] = pk2(fv.x, fv.y);
    }
    cq = __builtin_amdgcn_wmma_f32_16x16x32_bf16(false, aq.v, false, xb.v, (short)0, cq, false, false);
    ck = __builtin_amdgcn_wmma_f32_16x16x32_bf16(false, ak.v, false, xb.v, (short)0, ck, false, false);
    cv = __builtin_amdgcn_wmma_f32_16x16x32_bf16(false, av.v, false, xb.v, (short)0, cv, false, false);
  }
  // Epilogue. This lane's 8 D rows are o = o_base + half*8 + r (contiguous,
  // within one 64-block) -> one b128 store for Qtr and Ktr.
  const int o0 = o_base + half * 8;
  const int h  = o0 >> 6, w0 = o0 & 63;
  const int bh = b * HH + h;
  v4u qv, kv;
  #pragma unroll
  for (int j = 0; j < 4; ++j) {
    int oa = o0 + 2 * j, ob = oa + 1;
    qv[j] = pk2((cq[2*j] + bq[oa]) * 0.125f, (cq[2*j+1] + bq[ob]) * 0.125f);
    kv[j] = pk2( ck[2*j] + bk[oa],            ck[2*j+1] + bk[ob]);
  }
  *(v4u*)(Qtr + ((size_t)bh * SS + s) * WD + w0) = qv;
  *(v4u*)(Ktr + ((size_t)bh * SS + s) * WD + w0) = kv;
  #pragma unroll
  for (int r = 0; r < 8; ++r) {
    int o = o0 + r;
    Vb[((size_t)bh * WD + (o & 63)) * SS + s] = bf1(cv[r] + bv[o]);
  }
}

// ---------------------------------------------------------------------------
// Attention: softmax over s per output column t (axis=-2 of [b,h,s,t]).
// One block per (bh, 16-column t-tile). 8 waves each own 512 s.
// GEMM1: D1[m=s, n=t] = Qtr_tile * Ktr_tile  (K = w = 64, two WMMAs)
// GEMM2: acc[m=w, n=t] += V_tile * P_tile    (K = s = 32 per iteration)
// ---------------------------------------------------------------------------
__global__ __launch_bounds__(256)
void attn_kernel(const unsigned short* __restrict__ Qtr,
                 const unsigned short* __restrict__ Ktr,
                 const unsigned short* __restrict__ Vb,
                 float* __restrict__ out)
{
  __shared__ unsigned short pbuf[8][16][40];   // per-wave P^T tile, 80B rows
  __shared__ float mbuf[8][16], lbuf[8][16];
  __shared__ float Oacc[64][17];
  __shared__ float ltot[16];

  const int tid  = threadIdx.x;
  const int wave = tid >> 5, lane = tid & 31;
  const int bh     = blockIdx.x >> 8;
  const int t_base = (blockIdx.x & 255) << 4;
  const int half = lane >> 4, l16 = lane & 15;
  const int t = t_base + l16;

  // B1 fragments from Ktr: constant for the whole kernel (reused 256x).
  // Lane's 16 K values (w = c*32 + half*16 + 0..15) are contiguous: 2xb128.
  Frag b1[2];
  {
    const unsigned short* kt = Ktr + ((size_t)bh * SS + t) * WD;
    #pragma unroll
    for (int c = 0; c < 2; ++c) {
      b1[c].q[0] = *(const v4u*)(kt + c * 32 + half * 16);
      b1[c].q[1] = *(const v4u*)(kt + c * 32 + half * 16 + 8);
    }
  }

  const unsigned short* qb = Qtr + (size_t)bh * SS * WD;
  const unsigned short* vb = Vb  + (size_t)bh * WD * SS;

  const int s0 = wave * (SS / 8);
  float m = -1e30f, l = 0.f;
  v8f acc[4] = {v8f{}, v8f{}, v8f{}, v8f{}};

  for (int it = 0; it < 16; ++it) {              // 32 s per iteration
    int sbase = s0 + it * 32;
    __builtin_prefetch(qb + (size_t)(sbase + 32 + l16) * WD, 0, 0);

    // ---- GEMM1: two 16x16 score sub-tiles, K = w = 64 ----
    v8f d[2];
    #pragma unroll
    for (int sub = 0; sub < 2; ++sub) {
      const unsigned short* qp = qb + (size_t)(sbase + sub * 16 + l16) * WD;
      v8f dd = {};
      #pragma unroll
      for (int ch = 0; ch < 2; ++ch) {
        // A frag: u[0..7] at w = ch*32 + half*16 + 0..15 (contiguous)
        Frag a1;
        a1.q[0] = *(const v4u*)(qp + ch * 32 + half * 16);
        a1.q[1] = *(const v4u*)(qp + ch * 32 + half * 16 + 8);
        dd = __builtin_amdgcn_wmma_f32_16x16x32_bf16(false, a1.v, false, b1[ch].v, (short)0, dd, false, false);
      }
      d[sub] = dd;
    }

    // ---- online softmax over these 32 s (per column t) ----
    float tmax = -1e30f;
    #pragma unroll
    for (int sub = 0; sub < 2; ++sub)
      #pragma unroll
      for (int r = 0; r < 8; ++r) tmax = fmaxf(tmax, d[sub][r]);
    tmax = fmaxf(tmax, __shfl_xor(tmax, 16, 32));
    float mnew = fmaxf(m, tmax);
    float sc = __expf(m - mnew);
    l *= sc;
    #pragma unroll
    for (int wt = 0; wt < 4; ++wt)
      #pragma unroll
      for (int r = 0; r < 8; ++r) acc[wt][r] *= sc;

    float psum = 0.f;
    #pragma unroll
    for (int sub = 0; sub < 2; ++sub)
      #pragma unroll
      for (int j = 0; j < 4; ++j) {
        float p0 = __expf(d[sub][2*j]   - mnew);
        float p1 = __expf(d[sub][2*j+1] - mnew);
        psum += p0 + p1;
        // P[t][s]: lane's rows s = sub*16 + half*8 + 2j (+1): packed pair
        *(unsigned*)&pbuf[wave][l16][sub * 16 + half * 8 + 2 * j] = pk2(p0, p1);
      }
    psum += __shfl_xor(psum, 16, 32);
    l += psum;
    m = mnew;

    // ---- B2 fragment from LDS: K = s = half*16 + 0..15, contiguous ----
    Frag b2;
    b2.q[0] = *(const v4u*)&pbuf[wave][l16][half * 16];
    b2.q[1] = *(const v4u*)&pbuf[wave][l16][half * 16 + 8];

    // ---- GEMM2: acc[w, t] += V[w, s] * P[s, t], 4 w-tiles ----
    #pragma unroll
    for (int wt = 0; wt < 4; ++wt) {
      const unsigned short* vp = vb + (size_t)(wt * 16 + l16) * SS + sbase;
      Frag a2;
      a2.q[0] = *(const v4u*)(vp + half * 16);
      a2.q[1] = *(const v4u*)(vp + half * 16 + 8);
      acc[wt] = __builtin_amdgcn_wmma_f32_16x16x32_bf16(false, a2.v, false, b2.v, (short)0, acc[wt], false, false);
    }
  }

  // ---- cross-wave flash combine ----
  if (half == 0) { mbuf[wave][l16] = m; lbuf[wave][l16] = l; }
  __syncthreads();
  for (int i = tid; i < 64 * 17; i += 256) ((float*)Oacc)[i] = 0.f;
  __syncthreads();

  float gm = -1e30f;
  #pragma unroll
  for (int wv = 0; wv < 8; ++wv) gm = fmaxf(gm, mbuf[wv][l16]);
  float lt = 0.f;
  #pragma unroll
  for (int wv = 0; wv < 8; ++wv) lt += lbuf[wv][l16] * __expf(mbuf[wv][l16] - gm);
  float alpha = __expf(m - gm);

  #pragma unroll
  for (int wt = 0; wt < 4; ++wt)
    #pragma unroll
    for (int r = 0; r < 8; ++r)
      atomicAdd(&Oacc[wt * 16 + r + half * 8][l16], acc[wt][r] * alpha);
  if (wave == 0 && half == 0) ltot[l16] = lt;
  __syncthreads();

  float* ob = out + (size_t)bh * 64 * SS + t_base;   // out[b, o, s] f32
  for (int i = tid; i < 1024; i += 256) {
    int w = i >> 4, tt = i & 15;
    ob[(size_t)w * SS + tt] = Oacc[w][tt] / ltot[tt];
  }
}

// ---------------------------------------------------------------------------
extern "C" void kernel_launch(void* const* d_in, const int* in_sizes, int n_in,
                              void* d_out, int out_size, void* d_ws, size_t ws_size,
                              hipStream_t stream) {
  const float* X  = (const float*)d_in[0];
  const float* Wq = (const float*)d_in[1];
  const float* bq = (const float*)d_in[2];
  const float* Wk = (const float*)d_in[3];
  const float* bk = (const float*)d_in[4];
  const float* Wv = (const float*)d_in[5];
  const float* bv = (const float*)d_in[6];
  // n_heads (d_in[7]) is fixed to 4 in this problem.

  const size_t elems = (size_t)BB * HH * SS * WD;   // 4 Mi bf16 elements each
  unsigned short* qtr = (unsigned short*)d_ws;
  unsigned short* ktr = qtr + elems;
  unsigned short* vbp = ktr + elems;

  proj_kernel<<<2048, 256, 0, stream>>>(X, Wq, bq, Wk, bk, Wv, bv, qtr, ktr, vbp);
  attn_kernel<<<4096, 256, 0, stream>>>(qtr, ktr, vbp, (float*)d_out);
}
